// MHSA_3521873183477
// MI455X (gfx1250) — matrix operand order repaired and checked by reading, not compiled, not run on hardware
//
#include <hip/hip_runtime.h>
#include <hip/hip_bf16.h>

// ---- types -----------------------------------------------------------------
typedef __attribute__((ext_vector_type(16))) __bf16 bf16x16;
typedef __attribute__((ext_vector_type(8)))  __bf16 bf16x8;
typedef __attribute__((ext_vector_type(8)))  float  f32x8;

#define NHEADS 16
#define SEQ    2048
#define DIM    1024
#define HD     64          // head dim
#define BATCH  2

#define WMMA_BF16(a, b, c) __builtin_amdgcn_wmma_f32_16x16x32_bf16( \
    false, (a), false, (b), (short)0, (c), false, false)

// ---- WMMA operand loaders (wave32, 16x16x32 bf16) --------------------------
// A operand: 16(M) x 32(K), row-major source with leading dim ld.
// lanes 0-15: row M=lane,   elems 0..15 = K 0..15
// lanes16-31: row M=lane-16,elems 0..15 = K 16..31
__device__ __forceinline__ bf16x16 load_a(const __bf16* __restrict__ p, int ld,
                                          int row0, int k0, int lane) {
    int m    = lane & 15;
    int koff = (lane >> 4) << 4;                 // 0 or 16
    const __bf16* base = p + (size_t)(row0 + m) * ld + k0;
    bf16x8 lo = *(const bf16x8*)(base + koff);
    bf16x8 hi = *(const bf16x8*)(base + 8 + koff);
    return __builtin_shufflevector(lo, hi, 0,1,2,3,4,5,6,7,8,9,10,11,12,13,14,15);
}

// B operand: 32(K) x 16(N); source is row-major "N rows x K cols" (i.e. B^T),
// so a B column (fixed N, K=0..31) is contiguous in memory.
__device__ __forceinline__ bf16x16 load_b(const __bf16* __restrict__ p, int ld,
                                          int n0, int k0, int lane) {
    int n  = lane & 15;
    int kk = (lane >> 4) << 4;                   // 0 or 16
    return *(const bf16x16*)(p + (size_t)(n0 + n) * ld + k0 + kk);
}

// ---- f32 -> bf16 conversion ------------------------------------------------
__global__ void cvt_f32_bf16(const float* __restrict__ src,
                             __bf16* __restrict__ dst, int n) {
    int i = blockIdx.x * blockDim.x + threadIdx.x;
    if (i < n) dst[i] = (__bf16)src[i];
}

// ---- QKV projection: y = x @ Wqkv^T + bqkv, scattered to q/k/v -------------
// x: (4096,1024) bf16, W: (3072,1024) bf16 row-major. Wave tile: 32x64.
__global__ __launch_bounds__(256) void qkv_gemm(const __bf16* __restrict__ xb,
                                                const __bf16* __restrict__ wb,
                                                const float*  __restrict__ bias,
                                                __bf16* __restrict__ q,
                                                __bf16* __restrict__ k,
                                                __bf16* __restrict__ v) {
    int wid  = (blockIdx.x * blockDim.x + threadIdx.x) >> 5;
    int lane = threadIdx.x & 31;
    int mt = wid % 128;                 // 4096/32 M tiles
    int ng = wid / 128;                 // 3072/64 N groups
    int row0 = mt * 32;
    int n0   = ng * 64;

    f32x8 acc[2][4] = {};
    for (int k0 = 0; k0 < DIM; k0 += 32) {
        bf16x16 a0 = load_a(xb, DIM, row0,      k0, lane);
        bf16x16 a1 = load_a(xb, DIM, row0 + 16, k0, lane);
        #pragma unroll
        for (int t = 0; t < 4; ++t) {
            bf16x16 b = load_b(wb, DIM, n0 + t * 16, k0, lane);
            acc[0][t] = WMMA_BF16(a0, b, acc[0][t]);
            acc[1][t] = WMMA_BF16(a1, b, acc[1][t]);
        }
    }
    // epilogue: bias + scatter into (b,h,n,d) with o = h*192 + d*3 + s
    int colL  = lane & 15;
    int half8 = (lane >> 4) * 8;
    #pragma unroll
    for (int t = 0; t < 4; ++t) {
        int o  = n0 + t * 16 + colL;
        int s  = o % 3;
        int dd = (o / 3) & 63;
        int hh = o / 192;
        __bf16* dst = (s == 0) ? q : ((s == 1) ? k : v);
        float bv = bias[o];
        #pragma unroll
        for (int mi = 0; mi < 2; ++mi) {
            #pragma unroll
            for (int j = 0; j < 8; ++j) {
                int mrow = row0 + mi * 16 + j + half8;
                int bb = mrow >> 11;
                int nn = mrow & (SEQ - 1);
                dst[(((size_t)bb * NHEADS + hh) * SEQ + nn) * HD + dd] =
                    (__bf16)(acc[mi][t][j] + bv);
            }
        }
    }
}

// ---- Flash attention: per block = one (b,h), 128 q rows (8 waves x 16) -----
__global__ __launch_bounds__(256) void attn_kernel(const __bf16* __restrict__ q,
                                                   const __bf16* __restrict__ kmat,
                                                   const __bf16* __restrict__ vmat,
                                                   __bf16* __restrict__ ao) {
    __shared__ __align__(32) __bf16 Ks[32 * HD];        // K chunk, row-major
    __shared__ __align__(32) __bf16 Vt[HD * 32];        // V chunk, transposed (d-major)
    __shared__ __align__(32) __bf16 Ps[8][16 * 32];     // per-wave P staging

    int tid  = threadIdx.x;
    int w    = tid >> 5;
    int lane = tid & 31;
    int bh   = blockIdx.x >> 4;                         // 0..31
    int qt   = blockIdx.x & 15;
    const __bf16* qh = q    + (size_t)bh * SEQ * HD;
    const __bf16* kh = kmat + (size_t)bh * SEQ * HD;
    const __bf16* vh = vmat + (size_t)bh * SEQ * HD;
    int qrow0 = qt * 128 + w * 16;

    bf16x16 aq0 = load_a(qh, HD, qrow0, 0,  lane);      // d 0..31
    bf16x16 aq1 = load_a(qh, HD, qrow0, 32, lane);      // d 32..63

    f32x8 acc[4] = {};
    float rmax[8], rsum[8];
    #pragma unroll
    for (int j = 0; j < 8; ++j) { rmax[j] = -1e30f; rsum[j] = 0.0f; }

    int colL  = lane & 15;
    int khalf = (lane >> 4) * 16;
    int half8 = (lane >> 4) * 8;
    int koff  = (lane >> 4) * 16;

    // per-thread staging coordinates (32 rows x 64 cols, 8 bf16 per thread)
    int sr = tid >> 3;                 // 0..31 (kv row in chunk)
    int sc = (tid & 7) * 8;            // 0..56 (d)
    unsigned ldsK = (unsigned)(size_t)&Ks[sr * HD + sc];   // LDS byte offset

    for (int kc = 0; kc < SEQ; kc += 32) {
        // --- K chunk: CDNA5 async copy global -> LDS (ASYNCcnt-tracked) ---
        {
            unsigned long long gk =
                (unsigned long long)(kh + (size_t)(kc + sr) * HD + sc);
            asm volatile("global_load_async_to_lds_b128 %0, %1, off"
                         :: "v"(ldsK), "v"(gk) : "memory");
        }
        // --- V chunk: load to regs, scatter-transpose into LDS ---
        {
            bf16x8 v8 = *(const bf16x8*)(vh + (size_t)(kc + sr) * HD + sc);
            #pragma unroll
            for (int i = 0; i < 8; ++i) Vt[(sc + i) * 32 + sr] = v8[i];
        }
        asm volatile("s_wait_asynccnt 0x0" ::: "memory");
        __syncthreads();

        // --- S = Q K^T : 16 x 32 (two 16-col sub-tiles, contraction d=64) ---
        f32x8 s0 = {}, s1 = {};
        {
            bf16x16 b00 = *(const bf16x16*)&Ks[colL * HD + 0  + khalf];
            bf16x16 b01 = *(const bf16x16*)&Ks[colL * HD + 32 + khalf];
            s0 = WMMA_BF16(aq0, b00, s0);
            s0 = WMMA_BF16(aq1, b01, s0);
            bf16x16 b10 = *(const bf16x16*)&Ks[(16 + colL) * HD + 0  + khalf];
            bf16x16 b11 = *(const bf16x16*)&Ks[(16 + colL) * HD + 32 + khalf];
            s1 = WMMA_BF16(aq0, b10, s1);
            s1 = WMMA_BF16(aq1, b11, s1);
        }

        // --- online softmax (each row lives in one 16-lane half) ---
        #pragma unroll
        for (int j = 0; j < 8; ++j) {
            float mloc = fmaxf(s0[j], s1[j]);
            mloc = fmaxf(mloc, __shfl_xor(mloc, 1, 32));
            mloc = fmaxf(mloc, __shfl_xor(mloc, 2, 32));
            mloc = fmaxf(mloc, __shfl_xor(mloc, 4, 32));
            mloc = fmaxf(mloc, __shfl_xor(mloc, 8, 32));
            float mnew = fmaxf(rmax[j], mloc);
            float corr = __expf(rmax[j] - mnew);
            float p0 = __expf(s0[j] - mnew);
            float p1 = __expf(s1[j] - mnew);
            float ls = p0 + p1;
            ls += __shfl_xor(ls, 1, 32);
            ls += __shfl_xor(ls, 2, 32);
            ls += __shfl_xor(ls, 4, 32);
            ls += __shfl_xor(ls, 8, 32);
            rsum[j] = rsum[j] * corr + ls;
            rmax[j] = mnew;
            #pragma unroll
            for (int t = 0; t < 4; ++t) acc[t][j] = acc[t][j] * corr;
            int rowP = j + half8;
            Ps[w][rowP * 32 + colL]      = (__bf16)p0;
            Ps[w][rowP * 32 + 16 + colL] = (__bf16)p1;
        }

        // --- O += P * V : A = P (via LDS re-layout), B = Vt (contiguous) ---
        {
            bf16x8 plo = *(const bf16x8*)&Ps[w][colL * 32 + koff];
            bf16x8 phi = *(const bf16x8*)&Ps[w][colL * 32 + 8 + koff];
            bf16x16 pa = __builtin_shufflevector(plo, phi,
                0,1,2,3,4,5,6,7,8,9,10,11,12,13,14,15);
            #pragma unroll
            for (int t = 0; t < 4; ++t) {
                bf16x16 bv = *(const bf16x16*)&Vt[(t * 16 + colL) * 32 + khalf];
                acc[t] = WMMA_BF16(pa, bv, acc[t]);
            }
        }
        __syncthreads();
    }

    // --- finalize: /rowsum then /sqrt(dim)=32 (reference scales after softmax)
    int bb = bh >> 4, hh = bh & 15;
    #pragma unroll
    for (int j = 0; j < 8; ++j) {
        float inv  = 1.0f / (rsum[j] * 32.0f);
        int   nrow = qrow0 + j + half8;
        size_t base = ((size_t)bb * SEQ + nrow) * DIM + hh * HD;
        #pragma unroll
        for (int t = 0; t < 4; ++t)
            ao[base + t * 16 + colL] = (__bf16)(acc[t][j] * inv);
    }
}

// ---- output projection: out = ao @ Wproj^T + bproj (f32 out) ---------------
__global__ __launch_bounds__(256) void proj_gemm(const __bf16* __restrict__ a,
                                                 const __bf16* __restrict__ wb,
                                                 const float*  __restrict__ bias,
                                                 float* __restrict__ out) {
    int wid  = (blockIdx.x * blockDim.x + threadIdx.x) >> 5;
    int lane = threadIdx.x & 31;
    int mt = wid & 127;                 // 128 M tiles of 32
    int ng = wid >> 7;                  // 16 N groups of 64
    int row0 = mt * 32;
    int n0   = ng * 64;

    f32x8 acc[2][4] = {};
    for (int k0 = 0; k0 < DIM; k0 += 32) {
        bf16x16 a0 = load_a(a, DIM, row0,      k0, lane);
        bf16x16 a1 = load_a(a, DIM, row0 + 16, k0, lane);
        #pragma unroll
        for (int t = 0; t < 4; ++t) {
            bf16x16 b = load_b(wb, DIM, n0 + t * 16, k0, lane);
            acc[0][t] = WMMA_BF16(a0, b, acc[0][t]);
            acc[1][t] = WMMA_BF16(a1, b, acc[1][t]);
        }
    }
    int colL  = lane & 15;
    int half8 = (lane >> 4) * 8;
    #pragma unroll
    for (int t = 0; t < 4; ++t) {
        int o = n0 + t * 16 + colL;
        float bv = bias[o];
        #pragma unroll
        for (int mi = 0; mi < 2; ++mi) {
            #pragma unroll
            for (int j = 0; j < 8; ++j) {
                int mrow = row0 + mi * 16 + j + half8;
                out[(size_t)mrow * DIM + o] = acc[mi][t][j] + bv;
            }
        }
    }
}

// ---- host launcher ---------------------------------------------------------
extern "C" void kernel_launch(void* const* d_in, const int* in_sizes, int n_in,
                              void* d_out, int out_size, void* d_ws, size_t ws_size,
                              hipStream_t stream) {
    const float* x     = (const float*)d_in[0];   // (2,2048,1024)
    const float* Wqkv  = (const float*)d_in[1];   // (3072,1024)
    const float* bqkv  = (const float*)d_in[2];   // (3072,)
    const float* Wproj = (const float*)d_in[3];   // (1024,1024)
    const float* bproj = (const float*)d_in[4];   // (1024,)
    float* out = (float*)d_out;

    const int NX  = BATCH * SEQ * DIM;            // 4194304
    const int NWQ = 3 * DIM * DIM;                // 3145728
    const int NWP = DIM * DIM;                    // 1048576

    char* ws = (char*)d_ws;
    const size_t MB = 1024u * 1024u;
    __bf16* xb  = (__bf16*)(ws);                  //  8 MB
    __bf16* wqb = (__bf16*)(ws +  8 * MB);        //  6 MB
    __bf16* wpb = (__bf16*)(ws + 14 * MB);        //  2 MB
    __bf16* qb  = (__bf16*)(ws + 16 * MB);        //  8 MB
    __bf16* kb  = (__bf16*)(ws + 24 * MB);        //  8 MB
    __bf16* vb  = (__bf16*)(ws + 32 * MB);        //  8 MB
    __bf16* aob = (__bf16*)(ws + 40 * MB);        //  8 MB (total 48 MB)

    cvt_f32_bf16<<<(NX  + 255) / 256, 256, 0, stream>>>(x,     xb,  NX);
    cvt_f32_bf16<<<(NWQ + 255) / 256, 256, 0, stream>>>(Wqkv,  wqb, NWQ);
    cvt_f32_bf16<<<(NWP + 255) / 256, 256, 0, stream>>>(Wproj, wpb, NWP);

    // 4096/32=128 M-tiles, 3072/64=48 N-groups -> 6144 waves -> 768 blocks
    qkv_gemm<<<768, 256, 0, stream>>>(xb, wqb, bqkv, qb, kb, vb);

    // 32 (b,h) pairs x 16 q-tiles of 128 rows
    attn_kernel<<<512, 256, 0, stream>>>(qb, kb, vb, aob);

    // 128 M-tiles x 16 N-groups -> 2048 waves -> 256 blocks
    proj_gemm<<<256, 256, 0, stream>>>(aob, wpb, bproj, out);
}
